// ClassificationGCNFromPYG_25013889532263
// MI455X (gfx1250) — compile-verified, matrix operand order checked by reading, not run on hardware
//
#include <hip/hip_runtime.h>
#include <stdint.h>

// ---------------------------------------------------------------------------
// GCN (PyG-style) forward for MI455X / gfx1250, wave32.
// GEMMs: v_wmma_f32_16x16x32_bf16 with TDM (tensor_load_to_lds) A-tile staging
// and pre-swizzled global B fragments.
// ---------------------------------------------------------------------------

typedef __attribute__((ext_vector_type(16))) __bf16   v16bf;
typedef __attribute__((ext_vector_type(8)))  float    v8f;
typedef __attribute__((ext_vector_type(4)))  unsigned v4u;
typedef __attribute__((ext_vector_type(8)))  unsigned v8u;

__global__ void k_fill(float* __restrict__ p, int n, float v) {
  int i = blockIdx.x * blockDim.x + threadIdx.x;
  if (i < n) p[i] = v;
}

__global__ void k_deg(const int* __restrict__ dst, float* __restrict__ deg, int e) {
  int i = blockIdx.x * blockDim.x + threadIdx.x;
  if (i < e) atomicAdd(deg + dst[i], 1.0f);
}

__global__ void k_rsqrt_inplace(float* __restrict__ p, int n) {
  int i = blockIdx.x * blockDim.x + threadIdx.x;
  if (i < n) p[i] = rsqrtf(p[i]);
}

// Z[i, c] = (c < realc) ? b[c] : 0
__global__ void k_bias(float* __restrict__ Z, const float* __restrict__ b,
                       int n, int nc, int realc) {
  long long i = (long long)blockIdx.x * blockDim.x + threadIdx.x;
  if (i < (long long)n * nc) {
    int c = (int)(i % nc);
    Z[i] = (c < realc) ? b[c] : 0.0f;
  }
}

// ---------------------------------------------------------------------------
// Pre-swizzle W[K x realN] (f32, row-major) into per-lane WMMA B-fragment
// order, zero-padded: Wz[p][g][lane][h], h = 16 contiguous bf16 per lane.
// 16-bit B-fragment layout (mirrors A): lane&15 = column within group,
// lane>>4 selects K-halves {koff..koff+7, 16+koff..16+koff+7}, koff=(lane>>4)*8.
// ---------------------------------------------------------------------------
__global__ void k_wswz(const float* __restrict__ W, __bf16* __restrict__ Wz,
                       int K, int realN, int NG) {
  int idx = blockIdx.x * blockDim.x + threadIdx.x;
  int total = ((K + 31) / 32) * NG * 512;
  if (idx >= total) return;
  int h      = idx & 15;
  int lane   = (idx >> 4) & 31;
  int g      = (idx >> 9) % NG;
  int p      = idx / (NG * 512);
  int laneHi = lane >> 4;
  int n      = g * 16 + (lane & 15);
  int kk     = (h < 8) ? (laneHi * 8 + h) : (16 + laneHi * 8 + (h - 8));
  int gk     = p * 32 + kk;
  float v = (gk < K && n < realN) ? W[(size_t)gk * realN + n] : 0.0f;
  Wz[idx] = (__bf16)v;
}

// ---------------------------------------------------------------------------
// TDM: async DMA of a (tile_dim1=128 rows) x (tile_dim0=32 cols) f32 tile into
// LDS. Descriptor anchored at the tile start; tensor_dim0/1 = remaining
// extents so OOB reads return zero (handles K tail and M tail).
// D# group0: count=1 | lds_addr | global_addr[31:0] | addr[56:32] + type=2.
// D# group1: data_size=4B, tensor_dim0/1, tile_dim0=32, tile_dim1=128, stride.
// ---------------------------------------------------------------------------
__device__ __forceinline__ void tdm_load_tile(const float* gsrc, void* lds,
                                              unsigned rem_k, unsigned rem_m,
                                              unsigned strideK) {
  unsigned long long ga = (unsigned long long)(uintptr_t)gsrc;
  v4u g0;
  g0.x = 1u;                                     // count=1, user mode
  g0.y = (unsigned)(uintptr_t)lds;               // lds_addr (low 32 bits of flat)
  g0.z = (unsigned)ga;                           // global_addr[31:0]
  g0.w = ((unsigned)(ga >> 32) & 0x01FFFFFFu)    // global_addr[56:32]
         | 0x80000000u;                          // type = 2 ("image")
  v8u g1;
  g1[0] = 2u << 16;                              // data_size = 2 -> 4 bytes
  g1[1] = (rem_k & 0xFFFFu) << 16;               // tensor_dim0[15:0]
  g1[2] = (rem_k >> 16) | ((rem_m & 0xFFFFu) << 16);  // dim0[31:16] | dim1[15:0]
  g1[3] = (rem_m >> 16) | (32u << 16);           // dim1[31:16] | tile_dim0 = 32
  g1[4] = 128u;                                  // tile_dim1 = 128, tile_dim2 = 0
  g1[5] = strideK;                               // tensor_dim0_stride[31:0]
  g1[6] = 0u;                                    // stride[47:32], dim1_stride lo
  g1[7] = 0u;
  asm volatile("tensor_load_to_lds %0, %1" :: "s"(g0), "s"(g1) : "memory");
}

// ---------------------------------------------------------------------------
// WMMA bf16 GEMM: H[M x NCOL] = X[M x K] @ W (pre-swizzled bf16 fragments).
// Block = 256 threads = 8 waves; wave w owns rows [blk*128 + w*16, +16).
// A tiles staged by TDM, double-buffered (2 x 16KB LDS); wave 0 drives the
// DMA pipeline with s_wait_tensorcnt, barriers publish tiles to all waves.
// ---------------------------------------------------------------------------
template <int NCOL>
__global__ void __launch_bounds__(256)
k_gemm_wmma(const float* __restrict__ X, const __bf16* __restrict__ Wz,
            float* __restrict__ H, int M, int K) {
  constexpr int NG = NCOL / 16;
  __shared__ float sX[2][128 * 32];

  const int tid  = threadIdx.x;
  const int lane = tid & 31;
  const int wave = tid >> 5;
  const int row_block = blockIdx.x * 128;
  const int niter = (K + 31) >> 5;
  const unsigned rem_m = (unsigned)(M - row_block);

  v8f zacc = {};
  v8f acc[NG];
#pragma unroll
  for (int g = 0; g < NG; ++g) acc[g] = zacc;

  if (wave == 0)
    tdm_load_tile(X + (size_t)row_block * K, (void*)&sX[0][0],
                  (unsigned)K, rem_m, (unsigned)K);

  const int koff = (lane >> 4) << 3;   // 0 or 8
  const float* srow_base = &sX[0][0] + (((wave << 4) + (lane & 15)) << 5);

  for (int it = 0; it < niter; ++it) {
    if (wave == 0) __builtin_amdgcn_s_wait_tensorcnt(0);  // tile `it` landed
    __syncthreads();                                      // publish to all waves
    if (wave == 0 && it + 1 < niter) {                    // overlap next DMA
      int k0n = (it + 1) * 32;
      tdm_load_tile(X + (size_t)row_block * K + k0n,
                    (void*)&sX[(it + 1) & 1][0],
                    (unsigned)(K - k0n), rem_m, (unsigned)K);
    }

    // A fragment: f32 at k = koff..koff+7 and 16+koff..16+koff+7 of this lane's row -> bf16.
    const float* srow = srow_base + (it & 1) * (128 * 32);
    union { v16bf v; __bf16 h[16]; } a;
#pragma unroll
    for (int j = 0; j < 8; ++j) a.h[j]     = (__bf16)srow[koff + j];
#pragma unroll
    for (int j = 0; j < 8; ++j) a.h[8 + j] = (__bf16)srow[16 + koff + j];

    // B fragments: contiguous 32B per lane from pre-swizzled global W.
    const __bf16* wp = Wz + (size_t)it * (NG * 512) + lane * 16;
    if (it + 1 < niter) __builtin_prefetch(wp + NG * 512, 0, 1);
#pragma unroll
    for (int g = 0; g < NG; ++g) {
      v16bf bfr = *(const v16bf*)(wp + g * 512);
      acc[g] = __builtin_amdgcn_wmma_f32_16x16x32_bf16(
          false, a.v, false, bfr, (short)0, acc[g], false, false);
    }
    __syncthreads();   // everyone done with tile `it` before it is overwritten
  }

  // C/D layout: VGPR r -> row (lane<16 ? r : 8+r), col = lane&15.
  const int srow_st = row_block + (wave << 4) + ((lane >> 4) << 3);
  const int col     = lane & 15;
#pragma unroll
  for (int g = 0; g < NG; ++g) {
#pragma unroll
    for (int r = 0; r < 8; ++r) {
      int m = srow_st + r;
      if (m < M) H[(size_t)m * NCOL + g * 16 + col] = acc[g][r];
    }
  }
}

// One wave per edge; lane handles 2 consecutive features (coalesced gather of
// the source row, 64 consecutive f32 atomics on the destination row).
__global__ void k_scatter(const float* __restrict__ H, float* __restrict__ Z,
                          const float* __restrict__ dinv,
                          const int* __restrict__ src, const int* __restrict__ dst,
                          int E, int nc) {
  int warp = (int)((blockIdx.x * (long long)blockDim.x + threadIdx.x) >> 5);
  int lane = threadIdx.x & 31;
  if (warp >= E) return;
  int s = src[warp];
  int d = dst[warp];
  float nrm = dinv[s] * dinv[d];
  int f = lane * 2;
  if (f < nc) {
    const float* hp = H + (size_t)s * nc + f;
    float* zp = Z + (size_t)d * nc + f;
    atomicAdd(zp,     hp[0] * nrm);
    atomicAdd(zp + 1, hp[1] * nrm);
  }
}

// Fused: Z += self-loop (H * dinv^2), optional ReLU, and column-sum partials
// for the PairNorm column mean (LDS reduce -> one global atomic per column).
template <int NC>
__global__ void k_self_relu_colsum(const float* __restrict__ H,
                                   float* __restrict__ Z,
                                   const float* __restrict__ dinv,
                                   float* __restrict__ colsum,
                                   int n, int do_relu) {
  __shared__ float scol[NC];
  int tid = threadIdx.x;
  if (tid < NC) scol[tid] = 0.0f;
  __syncthreads();

  long long idx = (long long)blockIdx.x * blockDim.x + tid;
  int col = (int)(idx % NC);
  float z = 0.0f;
  if (idx < (long long)n * NC) {
    int row = (int)(idx / NC);
    float di = dinv[row];
    z = Z[idx] + H[idx] * di * di;
    if (do_relu) z = fmaxf(z, 0.0f);
    Z[idx] = z;
  }
  atomicAdd(&scol[col], z);
  __syncthreads();
  if (tid < NC) atomicAdd(&colsum[tid], scol[tid]);
}

// PairNorm (PN-SI): subtract column mean, scale row to unit norm. One wave per
// row, 2 features per lane, full-wave __shfl_xor reduction (wave32).
template <int NC, bool FINAL>
__global__ void k_pairnorm(float* __restrict__ Z, const float* __restrict__ colsum,
                           int n, float invN, float* __restrict__ out, int outc) {
  int lane = threadIdx.x & 31;
  int row = (int)((blockIdx.x * (long long)blockDim.x + threadIdx.x) >> 5);
  if (row >= n) return;  // uniform across the wave

  int f = lane * 2;
  float v0 = 0.0f, v1 = 0.0f;
  float* zrow = Z + (size_t)row * NC;
  if (f < NC) {
    v0 = zrow[f]     - colsum[f]     * invN;
    v1 = zrow[f + 1] - colsum[f + 1] * invN;
  }
  float ss = v0 * v0 + v1 * v1;
#pragma unroll
  for (int o = 16; o >= 1; o >>= 1) ss += __shfl_xor(ss, o, 32);
  float sc = rsqrtf(1e-6f + ss);

  if (FINAL) {
    if (f < outc)     out[(size_t)row * outc + f]     = 1.0f / (1.0f + expf(-v0 * sc));
    if (f + 1 < outc) out[(size_t)row * outc + f + 1] = 1.0f / (1.0f + expf(-v1 * sc));
  } else {
    if (f < NC) {
      zrow[f]     = v0 * sc;
      zrow[f + 1] = v1 * sc;
    }
  }
}

// ---------------------------------------------------------------------------
extern "C" void kernel_launch(void* const* d_in, const int* in_sizes, int n_in,
                              void* d_out, int out_size, void* d_ws, size_t ws_size,
                              hipStream_t stream) {
  (void)n_in; (void)out_size; (void)ws_size;
  const float* X  = (const float*)d_in[0];
  const float* Ws[4] = {(const float*)d_in[1], (const float*)d_in[3],
                        (const float*)d_in[5], (const float*)d_in[7]};
  const float* bs[4] = {(const float*)d_in[2], (const float*)d_in[4],
                        (const float*)d_in[6], (const float*)d_in[8]};
  const int* adj = (const int*)d_in[9];

  const int F0 = 1433;
  const int Nn = in_sizes[0] / F0;       // 100000
  const int E  = in_sizes[9] / 2;        // 1600000
  const int* srcE = adj;
  const int* dstE = adj + E;
  float* out = (float*)d_out;

  // Workspace carve (256B aligned): dinv | colsum | bufH | bufZ | Wz
  uintptr_t p = ((uintptr_t)d_ws + 255) & ~(uintptr_t)255;
  float* dinv = (float*)p;                 p += (size_t)Nn * sizeof(float);
  p = (p + 255) & ~(uintptr_t)255;
  float* colsum = (float*)p;               p += 64 * sizeof(float);
  p = (p + 255) & ~(uintptr_t)255;
  float* bufH = (float*)p;                 p += (size_t)Nn * 64 * sizeof(float);
  p = (p + 255) & ~(uintptr_t)255;
  float* bufZ = (float*)p;                 p += (size_t)Nn * 64 * sizeof(float);
  p = (p + 255) & ~(uintptr_t)255;
  __bf16* Wz = (__bf16*)p;                 // up to 45 panels * 4 groups * 512

  const int T = 256;
  const float invN = 1.0f / (float)Nn;

  // Degrees with self-loops, then dinv = rsqrt(deg).
  k_fill<<<(Nn + T - 1) / T, T, 0, stream>>>(dinv, Nn, 1.0f);
  k_deg<<<(E + T - 1) / T, T, 0, stream>>>(dstE, dinv, E);
  k_rsqrt_inplace<<<(Nn + T - 1) / T, T, 0, stream>>>(dinv, Nn);

  const int gemm_blocks = (Nn + 127) / 128;
  const int scat_blocks = (int)(((long long)E * 32 + T - 1) / T);
  const int pn_blocks   = (Nn + 7) / 8;

  // Layers 0..2: width-64, ReLU + PairNorm.
  const float* cur = X;
  int K = F0;
  for (int l = 0; l < 3; ++l) {
    int panels = (K + 31) / 32;
    int wz_total = panels * 4 * 512;
    k_wswz<<<(wz_total + T - 1) / T, T, 0, stream>>>(Ws[l], Wz, K, 64, 4);
    k_gemm_wmma<64><<<gemm_blocks, T, 0, stream>>>(cur, Wz, bufH, Nn, K);
    k_bias<<<(int)(((long long)Nn * 64 + T - 1) / T), T, 0, stream>>>(bufZ, bs[l], Nn, 64, 64);
    k_fill<<<1, 64, 0, stream>>>(colsum, 64, 0.0f);
    k_scatter<<<scat_blocks, T, 0, stream>>>(bufH, bufZ, dinv, srcE, dstE, E, 64);
    k_self_relu_colsum<64><<<(int)(((long long)Nn * 64 + T - 1) / T), T, 0, stream>>>(
        bufH, bufZ, dinv, colsum, Nn, 1);
    k_pairnorm<64, false><<<pn_blocks, T, 0, stream>>>(bufZ, colsum, Nn, invN, nullptr, 0);
    cur = bufZ;
    K = 64;
  }

  // Layer 3: 7 real columns padded to 16; no ReLU; PairNorm + sigmoid -> d_out.
  {
    int panels = (64 + 31) / 32;  // 2
    int wz_total = panels * 1 * 512;
    k_wswz<<<(wz_total + T - 1) / T, T, 0, stream>>>(Ws[3], Wz, 64, 7, 1);
    k_gemm_wmma<16><<<gemm_blocks, T, 0, stream>>>(cur, Wz, bufH, Nn, 64);
    k_bias<<<(int)(((long long)Nn * 16 + T - 1) / T), T, 0, stream>>>(bufZ, bs[3], Nn, 16, 7);
    k_fill<<<1, 64, 0, stream>>>(colsum, 64, 0.0f);
    k_scatter<<<scat_blocks, T, 0, stream>>>(bufH, bufZ, dinv, srcE, dstE, E, 16);
    k_self_relu_colsum<16><<<(int)(((long long)Nn * 16 + T - 1) / T), T, 0, stream>>>(
        bufH, bufZ, dinv, colsum, Nn, 0);
    k_pairnorm<16, true><<<pn_blocks, T, 0, stream>>>(bufZ, colsum, Nn, invN, out, 7);
  }
}
